// TrueFalsePredictor_20461224198279
// MI455X (gfx1250) — compile-verified, matrix operand order checked by reading, not run on hardware
//
#include <hip/hip_runtime.h>

typedef __attribute__((ext_vector_type(16))) _Float16 v16h;
typedef __attribute__((ext_vector_type(8)))  _Float16 h8;
typedef __attribute__((ext_vector_type(4)))  _Float16 h4;
typedef __attribute__((ext_vector_type(8)))  float    v8f;
typedef __attribute__((ext_vector_type(4)))  unsigned int u32x4;
typedef __attribute__((ext_vector_type(8)))  int          i32x8;
typedef __attribute__((ext_vector_type(4)))  int          i32x4;

#define HF   128   // feature dim
#define TM   64    // triplets (rows) per block
#define AS   136   // A LDS row stride in halves (pad for banks, keeps 16B align)
#define HS   130   // hid LDS row stride in floats

#if __has_builtin(__builtin_amdgcn_tensor_load_to_lds) && \
    __has_builtin(__builtin_amdgcn_s_wait_tensorcnt)
#define USE_TDM 1
#else
#define USE_TDM 0
#endif

union __align__(16) Smem {
    struct {
        _Float16 A[TM * AS];        // 64 x 128 f16 gathered chunk (padded)  17408 B
        _Float16 B[16 * 128 * 8];   // K-chunk of packed W1: [kb][n][8]      32768 B
    } ab;
    float hid[TM * HS];             // 64 x 128 f32 hidden (padded)          33280 B
};

// ---- one-time: W1 [n=128][k=384] f32 -> B-fragment-major f16: BW[(k/8)][n][k%8]
__global__ __launch_bounds__(256) void pack_w1_kernel(const float* __restrict__ W1w,
                                                      _Float16* __restrict__ BW) {
    int e = blockIdx.x * 256 + threadIdx.x;
    if (e >= 128 * 384) return;
    int n = e / 384;
    int k = e % 384;
    BW[(size_t)(k >> 3) * 1024 + n * 8 + (k & 7)] = (_Float16)W1w[(size_t)n * 384 + k];
}

__global__ __launch_bounds__(256) void triplet_mlp_kernel(
    const float* __restrict__ h, const long long* __restrict__ trip,
    const _Float16* __restrict__ BW, const float* __restrict__ W1b,
    const float* __restrict__ W2w, const float* __restrict__ W2b,
    float* __restrict__ out, int T)
{
    __shared__ Smem sm;
    const int tid  = threadIdx.x;
    const int wave = tid >> 5;
    const int lane = tid & 31;
    const int ln   = lane & 15;   // N position within tile / M row within strip
    const int hfid = lane >> 4;   // half-wave id
    const int r16  = (wave >> 1) * 16;  // row strip base (0,16,32,48)
    const int c64  = (wave & 1) * 64;   // col strip base (0,64)
    const long long tileBase = (long long)blockIdx.x * TM;

    v8f acc[4];
#pragma unroll
    for (int nt = 0; nt < 4; ++nt)
#pragma unroll
        for (int j = 0; j < 8; ++j) acc[nt][j] = 0.0f;

    // gather-fill mapping: 4 threads per row, 32 cols each
    const int rowF = tid >> 2;
    const int sub  = tid & 3;

    // prefetch next tile's triplet indices (speculative; OOB is safe)
    __builtin_prefetch(trip + (tileBase + TM) * 3 + tid, 0, 1);

#pragma unroll
    for (int kc = 0; kc < 3; ++kc) {
        // ---- stage B: DMA 32 KB packed W1 K-slice into LDS via the TDM ----
#if USE_TDM
        if (tid < 32) {
            unsigned long long gaddr =
                (unsigned long long)(const void*)BW + (unsigned long long)kc * 32768ull;
            unsigned int ldsa = (unsigned int)(unsigned long long)(void*)sm.ab.B;
            u32x4 g0;
            g0[0] = 1u;                          // count=1, user descriptor
            g0[1] = ldsa;                        // lds_addr (bytes)
            g0[2] = (unsigned int)gaddr;         // global_addr[31:0]
            g0[3] = (unsigned int)((gaddr >> 32) & 0x1FFFFFFull) | 0x80000000u; // [56:32] | type=2
            i32x8 g1;
            g1[0] = 0x00010000;                  // workgroup_mask=0, data_size=2B
            g1[1] = 0x40000000;                  // tensor_dim0 = 16384 (lo16 in [63:48])
            g1[2] = 0x00010000;                  // tensor_dim0 hi=0, tensor_dim1 = 1
            g1[3] = 0x40000000;                  // tensor_dim1 hi=0, tile_dim0 = 16384
            g1[4] = 0x00000001;                  // tile_dim1 = 1, tile_dim2 = 0
            g1[5] = 16384;                       // tensor_dim0_stride lo
            g1[6] = 0;                           // stride hi / dim1_stride lo
            g1[7] = 0;
            i32x4 gz4;
            gz4[0] = 0; gz4[1] = 0; gz4[2] = 0; gz4[3] = 0;
            i32x8 gz8;
#pragma unroll
            for (int i = 0; i < 8; ++i) gz8[i] = 0;
            __builtin_amdgcn_tensor_load_to_lds(g0, g1, gz4, gz4, gz8, 0);
        }
#endif

        // ---- stage A: gather one h-row per (row, kc), convert f32 -> f16 ----
        long long t = tileBase + rowF;
        if (t > (long long)T - 1) t = T - 1;
        long long idx = trip[t * 3 + kc];
        const float*  src = h + (size_t)idx * HF + sub * 32;
        _Float16*     dA  = sm.ab.A + rowF * AS + sub * 32;
#pragma unroll
        for (int q = 0; q < 8; ++q) {
            float4 v = ((const float4*)src)[q];
            h4 p;
            p[0] = (_Float16)v.x; p[1] = (_Float16)v.y;
            p[2] = (_Float16)v.z; p[3] = (_Float16)v.w;
            ((h4*)dA)[q] = p;
        }

#if USE_TDM
        if (tid < 32) __builtin_amdgcn_s_wait_tensorcnt(0);
#else
        // ---- fallback stage B: coalesced 32 KB copy via VGPRs ----
        {
            const h8* srcB = (const h8*)(BW + (size_t)kc * 16384);
            h8*       dstB = (h8*)sm.ab.B;
#pragma unroll
            for (int r = 0; r < 8; ++r)
                dstB[r * 256 + tid] = srcB[r * 256 + tid];
        }
#endif
        __syncthreads();

        // ---- WMMA: K-chunk of 128 as 4 steps of K=32 ----
#pragma unroll
        for (int kb4 = 0; kb4 < 4; ++kb4) {
            const int kbase = kb4 * 32;
            // A fragment: lanes 0-15 K kbase..+7 & kbase+16..+23; lanes 16-31 shifted by 8
            const _Float16* pa = sm.ab.A + (r16 + ln) * AS + kbase + hfid * 8;
            h8 alo = *(const h8*)pa;
            h8 ahi = *(const h8*)(pa + 16);
            v16h a;
#pragma unroll
            for (int i = 0; i < 8; ++i) { a[i] = alo[i]; a[8 + i] = ahi[i]; }

#pragma unroll
            for (int nt = 0; nt < 4; ++nt) {
                const int n = c64 + nt * 16 + ln;
                const _Float16* pb = sm.ab.B + ((size_t)(kb4 * 4 + hfid) * 128 + n) * 8;
                h8 blo = *(const h8*)pb;
                h8 bhi = *(const h8*)(pb + 2 * 128 * 8);  // kbl + 2 -> K+16
                v16h b;
#pragma unroll
                for (int i = 0; i < 8; ++i) { b[i] = blo[i]; b[8 + i] = bhi[i]; }

                acc[nt] = __builtin_amdgcn_wmma_f32_16x16x32_f16(
                    /*neg_a=*/false, a, /*neg_b=*/false, b,
                    /*c_mod=*/(short)0, acc[nt],
                    /*reuse_a=*/false, /*reuse_b=*/false);
            }
        }
        __syncthreads();   // A/B consumed before next chunk overwrites
    }

    // ---- bias + ReLU, spill hid to LDS (union reuse) ----
#pragma unroll
    for (int nt = 0; nt < 4; ++nt) {
        const int n  = c64 + nt * 16 + ln;
        const float b1 = W1b[n];
#pragma unroll
        for (int j = 0; j < 8; ++j) {
            float v = acc[nt][j] + b1;
            v = v > 0.0f ? v : 0.0f;
            sm.hid[(r16 + j + 8 * hfid) * HS + n] = v;
        }
    }
    __syncthreads();

    // ---- second GEMM: [64 x 128] x [128 x 2] + bias ----
    if (tid < 128) {
        const int row = tid >> 1;
        const int o   = tid & 1;
        float a2 = W2b[o];
        const float* w2 = W2w + o * HF;
        const float* hr = sm.hid + row * HS;
#pragma unroll 8
        for (int n = 0; n < HF; ++n) a2 = fmaf(hr[n], w2[n], a2);
        long long t = tileBase + row;
        if (t < T) out[t * 2 + o] = a2;
    }
}

extern "C" void kernel_launch(void* const* d_in, const int* in_sizes, int n_in,
                              void* d_out, int out_size, void* d_ws, size_t ws_size,
                              hipStream_t stream) {
    const float*     h    = (const float*)d_in[0];
    const long long* trip = (const long long*)d_in[1];
    const float*     W1w  = (const float*)d_in[2];
    const float*     W1b  = (const float*)d_in[3];
    const float*     W2w  = (const float*)d_in[4];
    const float*     W2b  = (const float*)d_in[5];
    float*           out  = (float*)d_out;
    const int T = in_sizes[1] / 3;

    _Float16* BW = (_Float16*)d_ws;   // 384*128*2 = 96 KB packed f16 W1

    pack_w1_kernel<<<(128 * 384 + 255) / 256, 256, 0, stream>>>(W1w, BW);

    const int nblk = (T + TM - 1) / TM;
    triplet_mlp_kernel<<<nblk, 256, 0, stream>>>(h, trip, BW, W1b, W2w, W2b, out, T);
}
